// ChaosSSMCore_25907242729683
// MI455X (gfx1250) — compile-verified, matrix operand order checked by reading, not run on hardware
//
#include <hip/hip_runtime.h>
#include <math.h>
#include <stdint.h>

// ---------------------------------------------------------------------------
// ChaosSSMCore (diag mode) for MI455X / gfx1250, wave32 + WMMA + async-to-LDS.
//   B=4, S=4096, D=1024  -> M = B*S = 16384 rows.
// Pipeline:
//   1) f32 -> f16 convert of x and the 5 weight matrices (workspace)
//   2) fused WMMA GEMM:  update = sigmoid(x@Wsel^T) * tanh(x@Win^T)
//   3) WMMA GEMM + epilogue: decay = exp(-max(softplus(x@Wdelta^T),1e-4)*sigmoid(log_a))
//   4) WMMA GEMM + epilogue: gate = sigmoid(x@Wgate^T)
//   5) chunked parallel scan over S (64 chunks x 64 steps, 3 passes),
//      fusing outh = f16(gate * state)
//   6) WMMA GEMM: y = outh @ Wout^T  (f32 out -> d_out)
// GEMMs stage tiles with GLOBAL_LOAD_ASYNC_TO_LDS_B128 (ASYNCcnt), double
// buffered in LDS, fragments read back with ds_load_b128.
// ---------------------------------------------------------------------------

typedef __attribute__((ext_vector_type(16))) _Float16 v16h;
typedef __attribute__((ext_vector_type(8)))  _Float16 v8h;
typedef __attribute__((ext_vector_type(8)))  float    v8f;

#define DEV static __device__ __forceinline__

constexpr int kB   = 4;
constexpr int kS   = 4096;
constexpr int kD   = 1024;
constexpr int kM   = kB * kS;      // 16384 rows of the flattened [B*S, D] problem
constexpr int kNCH = 64;           // scan chunks along S
constexpr int kCL  = kS / kNCH;    // 64 steps per chunk
constexpr int kLP  = 40;           // LDS tile row pitch in f16 (80B: 16B-aligned,
                                   // 20 dwords -> spreads lanes over banks)

// ---------------- WMMA helpers --------------------------------------------

DEV v8f wmma_f16(v16h a, v16h b, v8f c) {
  // 8 args: (neg_a, A, neg_b, B, c_mod, C, reuse_a, reuse_b)
  return __builtin_amdgcn_wmma_f32_16x16x32_f16(false, a, false, b, (short)0, c,
                                                false, false);
}

// Read a 16x32 f16 fragment from an LDS tile [rows][kLP].
// ISA layout for 16-bit A/B 16x32: lanes 0-15 hold K=0..7,16..23 of row base+lane;
// lanes 16-31 hold K=8..15,24..31 of row base+lane-16. Two 16B ds loads per lane.
DEV v16h lds_frag(const _Float16* __restrict__ tile, int rbase, int lane) {
  const int r = rbase + (lane & 15);
  const int h = lane >> 4;
  const _Float16* q = tile + r * kLP + h * 8;
  v8h lo = *reinterpret_cast<const v8h*>(q);       // K seg [h*8, h*8+8)
  v8h hi = *reinterpret_cast<const v8h*>(q + 16);  // K seg [16+h*8, ...)
  v16h f;
#pragma unroll
  for (int i = 0; i < 8; ++i) { f[i] = lo[i]; f[i + 8] = hi[i]; }
  return f;
}

// Async-copy a ROWS x 32 f16 tile (row-major source, pitch kD) into LDS
// (pitch kLP). 16 bytes per lane per issue; tracked by ASYNCcnt.
// ROWS is compile-time so this is 1 (ROWS=64) or 2 (ROWS=128) straight-line
// async issues per thread -- no inner loop, no exec juggling.
template <int ROWS>
DEV void stage_tile_async(const _Float16* __restrict__ g, int grow0, int k0,
                          _Float16* l, int tid) {
  static_assert(ROWS == 64 || ROWS == 128, "tile rows");
#pragma unroll
  for (int c = tid; c < ROWS * 4; c += 256) {
    const int row = c >> 2;
    const int seg = c & 3;
    const _Float16* gp = g + (size_t)(grow0 + row) * kD + (k0 + seg * 8);
    const _Float16* lp = l + row * kLP + seg * 8;
    const unsigned lds_addr = (unsigned)(uintptr_t)lp;  // low 32b = LDS offset
    const unsigned long long gaddr = (unsigned long long)(uintptr_t)gp;
    asm volatile("global_load_async_to_lds_b128 %0, %1, off"
                 :: "v"(lds_addr), "v"(gaddr) : "memory");
  }
}

DEV void wait_async_barrier() {
  asm volatile("s_wait_asynccnt 0x0" ::: "memory");
  __syncthreads();
}

DEV float sigmoid_(float z) { return 1.0f / (1.0f + expf(-z)); }

// ---------------- convert f32 -> f16 --------------------------------------

__global__ void __launch_bounds__(256)
cvt_f32_to_f16(const float* __restrict__ in, _Float16* __restrict__ out, int n) {
  int i = blockIdx.x * blockDim.x + threadIdx.x;
  const int stride = gridDim.x * blockDim.x;
  for (; i < n; i += stride) out[i] = (_Float16)in[i];
}

// ---------------- fused update GEMM ----------------------------------------
// update = sigmoid(x @ Wsel^T) * tanh(x @ Win^T); x staged once for both GEMMs.
// Block = 256 threads = 8 waves; block tile 64(M) x 128(N); wave tile 32x32.

__global__ void __launch_bounds__(256)
proj_update_kernel(const _Float16* __restrict__ xh,
                   const _Float16* __restrict__ wsel,
                   const _Float16* __restrict__ win,
                   float* __restrict__ update) {
  alignas(16) __shared__ _Float16 As [2][64 * kLP];
  alignas(16) __shared__ _Float16 BsS[2][128 * kLP];
  alignas(16) __shared__ _Float16 BsI[2][128 * kLP];

  const int tid  = threadIdx.x;
  const int lane = tid & 31;
  const int wave = tid >> 5;
  const int rowB = blockIdx.y * 64;
  const int colB = blockIdx.x * 128;
  const int wm   = (wave & 1) * 32;    // wave row offset inside block tile
  const int wn   = (wave >> 1) * 32;   // wave col offset inside block tile

  stage_tile_async< 64>(xh,   rowB, 0, As [0], tid);
  stage_tile_async<128>(wsel, colB, 0, BsS[0], tid);
  stage_tile_async<128>(win,  colB, 0, BsI[0], tid);
  wait_async_barrier();

  v8f accS[2][2] = {{{0.f}}};
  v8f accI[2][2] = {{{0.f}}};

  int buf = 0;
  for (int k0 = 0; k0 < kD - 32; k0 += 32) {  // steady state: branch-free body
    stage_tile_async< 64>(xh,   rowB, k0 + 32, As [buf ^ 1], tid);
    stage_tile_async<128>(wsel, colB, k0 + 32, BsS[buf ^ 1], tid);
    stage_tile_async<128>(win,  colB, k0 + 32, BsI[buf ^ 1], tid);
    v16h a0 = lds_frag(As [buf], wm,      lane);
    v16h a1 = lds_frag(As [buf], wm + 16, lane);
    v16h s0 = lds_frag(BsS[buf], wn,      lane);
    v16h s1 = lds_frag(BsS[buf], wn + 16, lane);
    v16h i0 = lds_frag(BsI[buf], wn,      lane);
    v16h i1 = lds_frag(BsI[buf], wn + 16, lane);
    accS[0][0] = wmma_f16(a0, s0, accS[0][0]);
    accS[0][1] = wmma_f16(a0, s1, accS[0][1]);
    accS[1][0] = wmma_f16(a1, s0, accS[1][0]);
    accS[1][1] = wmma_f16(a1, s1, accS[1][1]);
    accI[0][0] = wmma_f16(a0, i0, accI[0][0]);
    accI[0][1] = wmma_f16(a0, i1, accI[0][1]);
    accI[1][0] = wmma_f16(a1, i0, accI[1][0]);
    accI[1][1] = wmma_f16(a1, i1, accI[1][1]);
    wait_async_barrier();
    buf ^= 1;
  }
  {  // peeled last K-step: nothing left to stage
    v16h a0 = lds_frag(As [buf], wm,      lane);
    v16h a1 = lds_frag(As [buf], wm + 16, lane);
    v16h s0 = lds_frag(BsS[buf], wn,      lane);
    v16h s1 = lds_frag(BsS[buf], wn + 16, lane);
    v16h i0 = lds_frag(BsI[buf], wn,      lane);
    v16h i1 = lds_frag(BsI[buf], wn + 16, lane);
    accS[0][0] = wmma_f16(a0, s0, accS[0][0]);
    accS[0][1] = wmma_f16(a0, s1, accS[0][1]);
    accS[1][0] = wmma_f16(a1, s0, accS[1][0]);
    accS[1][1] = wmma_f16(a1, s1, accS[1][1]);
    accI[0][0] = wmma_f16(a0, i0, accI[0][0]);
    accI[0][1] = wmma_f16(a0, i1, accI[0][1]);
    accI[1][0] = wmma_f16(a1, i0, accI[1][0]);
    accI[1][1] = wmma_f16(a1, i1, accI[1][1]);
  }

  const int n  = lane & 15;
  const int mh = (lane >> 4) * 8;   // C/D layout: m = r + 8*(lane>=16)
#pragma unroll
  for (int mi = 0; mi < 2; ++mi)
#pragma unroll
    for (int ni = 0; ni < 2; ++ni)
#pragma unroll
      for (int r = 0; r < 8; ++r) {
        const size_t row = (size_t)(rowB + wm + mi * 16 + mh + r);
        const size_t col = (size_t)(colB + wn + ni * 16 + n);
        update[row * kD + col] = sigmoid_(accS[mi][ni][r]) * tanhf(accI[mi][ni][r]);
      }
}

// ---------------- single-matrix GEMM with epilogue -------------------------
// MODE 0: identity (final y GEMM)   MODE 1: sigmoid (gate)
// MODE 2: decay epilogue (softplus/clamp/exp with a_base = sigmoid(log_a[col]))

template <int MODE>
__global__ void __launch_bounds__(256)
gemm16_kernel(const _Float16* __restrict__ A,
              const _Float16* __restrict__ W,
              const float* __restrict__ log_a,
              float* __restrict__ out) {
  (void)log_a;
  alignas(16) __shared__ _Float16 As[2][64 * kLP];
  alignas(16) __shared__ _Float16 Bs[2][128 * kLP];

  const int tid  = threadIdx.x;
  const int lane = tid & 31;
  const int wave = tid >> 5;
  const int rowB = blockIdx.y * 64;
  const int colB = blockIdx.x * 128;
  const int wm   = (wave & 1) * 32;
  const int wn   = (wave >> 1) * 32;

  stage_tile_async< 64>(A, rowB, 0, As[0], tid);
  stage_tile_async<128>(W, colB, 0, Bs[0], tid);
  wait_async_barrier();

  v8f acc[2][2] = {{{0.f}}};

  int buf = 0;
  for (int k0 = 0; k0 < kD - 32; k0 += 32) {
    stage_tile_async< 64>(A, rowB, k0 + 32, As[buf ^ 1], tid);
    stage_tile_async<128>(W, colB, k0 + 32, Bs[buf ^ 1], tid);
    v16h a0 = lds_frag(As[buf], wm,      lane);
    v16h a1 = lds_frag(As[buf], wm + 16, lane);
    v16h b0 = lds_frag(Bs[buf], wn,      lane);
    v16h b1 = lds_frag(Bs[buf], wn + 16, lane);
    acc[0][0] = wmma_f16(a0, b0, acc[0][0]);
    acc[0][1] = wmma_f16(a0, b1, acc[0][1]);
    acc[1][0] = wmma_f16(a1, b0, acc[1][0]);
    acc[1][1] = wmma_f16(a1, b1, acc[1][1]);
    wait_async_barrier();
    buf ^= 1;
  }
  {
    v16h a0 = lds_frag(As[buf], wm,      lane);
    v16h a1 = lds_frag(As[buf], wm + 16, lane);
    v16h b0 = lds_frag(Bs[buf], wn,      lane);
    v16h b1 = lds_frag(Bs[buf], wn + 16, lane);
    acc[0][0] = wmma_f16(a0, b0, acc[0][0]);
    acc[0][1] = wmma_f16(a0, b1, acc[0][1]);
    acc[1][0] = wmma_f16(a1, b0, acc[1][0]);
    acc[1][1] = wmma_f16(a1, b1, acc[1][1]);
  }

  const int n  = lane & 15;
  const int mh = (lane >> 4) * 8;
#pragma unroll
  for (int mi = 0; mi < 2; ++mi)
#pragma unroll
    for (int ni = 0; ni < 2; ++ni) {
      const int col = colB + wn + ni * 16 + n;
      float a_base = 0.0f;
      if (MODE == 2) a_base = sigmoid_(log_a[col]);
#pragma unroll
      for (int r = 0; r < 8; ++r) {
        const size_t row = (size_t)(rowB + wm + mi * 16 + mh + r);
        float z = acc[mi][ni][r];
        float v;
        if (MODE == 0) {
          v = z;
        } else if (MODE == 1) {
          v = sigmoid_(z);
        } else {
          float sp = (z > 20.0f) ? z : log1pf(expf(z));   // softplus, stable
          sp = fmaxf(sp, 1e-4f);
          v = expf(-sp * a_base);
        }
        out[row * kD + (size_t)col] = v;
      }
    }
}

// ---------------- chunked scan over S --------------------------------------
// state[s] = decay[s]*state[s-1] + update[s];   out = gate*state -> f16.
// Composition is associative: per chunk carry (A = prod decay, U = local scan end).

__global__ void __launch_bounds__(256)
scan_part1(const float* __restrict__ decay, const float* __restrict__ update,
           float* __restrict__ cA, float* __restrict__ cU) {
  const int tid  = threadIdx.x;
  const int dblk = blockIdx.x & 3;        // kD/256 = 4
  const int bc   = blockIdx.x >> 2;       // b*kNCH + ch
  const int d    = dblk * 256 + tid;
  const int b    = bc / kNCH;
  const int ch   = bc % kNCH;
  float A = 1.0f, U = 0.0f;
  size_t off = ((size_t)b * kS + (size_t)ch * kCL) * kD + d;
  for (int s = 0; s < kCL; ++s) {
    __builtin_prefetch(decay + off + 8 * kD, 0, 0);
    const float dec = decay[off];
    U = fmaf(dec, U, update[off]);
    A *= dec;
    off += kD;
  }
  const size_t ci = (size_t)bc * kD + d;
  cA[ci] = A;
  cU[ci] = U;
}

__global__ void __launch_bounds__(256)
scan_carry(const float* __restrict__ cA, const float* __restrict__ cU,
           float* __restrict__ cinit) {
  const int id = blockIdx.x * 256 + threadIdx.x;  // kB*kD = 4096 threads
  const int b = id / kD;
  const int d = id % kD;
  float st = 0.0f;
  for (int ch = 0; ch < kNCH; ++ch) {
    const size_t ci = (size_t)(b * kNCH + ch) * kD + d;
    cinit[ci] = st;                     // exclusive prefix: state entering chunk
    st = fmaf(cA[ci], st, cU[ci]);
  }
}

__global__ void __launch_bounds__(256)
scan_part2(const float* __restrict__ decay, const float* __restrict__ update,
           const float* __restrict__ gate, const float* __restrict__ cinit,
           _Float16* __restrict__ outh) {
  const int tid  = threadIdx.x;
  const int dblk = blockIdx.x & 3;
  const int bc   = blockIdx.x >> 2;
  const int d    = dblk * 256 + tid;
  const int b    = bc / kNCH;
  const int ch   = bc % kNCH;
  float st = cinit[(size_t)bc * kD + d];
  size_t off = ((size_t)b * kS + (size_t)ch * kCL) * kD + d;
  for (int s = 0; s < kCL; ++s) {
    __builtin_prefetch(gate + off + 8 * kD, 0, 0);
    st = fmaf(decay[off], st, update[off]);
    outh[off] = (_Float16)(gate[off] * st);
    off += kD;
  }
}

// ---------------- host launcher --------------------------------------------

extern "C" void kernel_launch(void* const* d_in, const int* in_sizes, int n_in,
                              void* d_out, int out_size, void* d_ws, size_t ws_size,
                              hipStream_t stream) {
  (void)in_sizes; (void)n_in; (void)out_size; (void)ws_size;

  const float* x       = (const float*)d_in[0];
  const float* W_in    = (const float*)d_in[1];
  const float* W_sel   = (const float*)d_in[2];
  const float* W_gate  = (const float*)d_in[3];
  const float* W_out   = (const float*)d_in[4];
  const float* W_delta = (const float*)d_in[5];
  const float* log_a   = (const float*)d_in[6];
  float* y = (float*)d_out;

  // workspace carve-up (256B aligned)
  char* p = (char*)d_ws;
  auto take = [&](size_t bytes) -> void* {
    void* r = (void*)p;
    p += (bytes + 255) & ~(size_t)255;
    return r;
  };
  const size_t MD = (size_t)kM * kD;
  const size_t DD = (size_t)kD * kD;
  _Float16* xh      = (_Float16*)take(MD * sizeof(_Float16));
  _Float16* wh_in   = (_Float16*)take(DD * sizeof(_Float16));
  _Float16* wh_sel  = (_Float16*)take(DD * sizeof(_Float16));
  _Float16* wh_gate = (_Float16*)take(DD * sizeof(_Float16));
  _Float16* wh_out  = (_Float16*)take(DD * sizeof(_Float16));
  _Float16* wh_del  = (_Float16*)take(DD * sizeof(_Float16));
  float* decay  = (float*)take(MD * sizeof(float));
  float* update = (float*)take(MD * sizeof(float));
  float* gate   = (float*)take(MD * sizeof(float));
  _Float16* outh = (_Float16*)take(MD * sizeof(_Float16));
  const size_t CN = (size_t)kB * kNCH * kD;
  float* cA    = (float*)take(CN * sizeof(float));
  float* cU    = (float*)take(CN * sizeof(float));
  float* cinit = (float*)take(CN * sizeof(float));

  // 1) precision conversion
  cvt_f32_to_f16<<<8192, 256, 0, stream>>>(x, xh, (int)MD);
  cvt_f32_to_f16<<<1024, 256, 0, stream>>>(W_in,    wh_in,   (int)DD);
  cvt_f32_to_f16<<<1024, 256, 0, stream>>>(W_sel,   wh_sel,  (int)DD);
  cvt_f32_to_f16<<<1024, 256, 0, stream>>>(W_gate,  wh_gate, (int)DD);
  cvt_f32_to_f16<<<1024, 256, 0, stream>>>(W_out,   wh_out,  (int)DD);
  cvt_f32_to_f16<<<1024, 256, 0, stream>>>(W_delta, wh_del,  (int)DD);

  // 2-4) projection GEMMs (block tile 64x128, grid = N/128 x M/64)
  dim3 ggrid(kD / 128, kM / 64);
  proj_update_kernel<<<ggrid, 256, 0, stream>>>(xh, wh_sel, wh_in, update);
  gemm16_kernel<2><<<ggrid, 256, 0, stream>>>(xh, wh_del, log_a, decay);
  gemm16_kernel<1><<<ggrid, 256, 0, stream>>>(xh, wh_gate, nullptr, gate);

  // 5) chunked scan: (B * NCH) chunks x (D/256) blocks of 256 lanes
  const int scan_blocks = kB * kNCH * (kD / 256);   // 1024
  scan_part1<<<scan_blocks, 256, 0, stream>>>(decay, update, cA, cU);
  scan_carry<<<(kB * kD) / 256, 256, 0, stream>>>(cA, cU, cinit);
  scan_part2<<<scan_blocks, 256, 0, stream>>>(decay, update, gate, cinit, outh);

  // 6) final GEMM into d_out
  gemm16_kernel<0><<<ggrid, 256, 0, stream>>>(outh, wh_out, nullptr, y);
}
